// MultiTokenAttention_61512521614069
// MI455X (gfx1250) — compile-verified, matrix operand order checked by reading, not run on hardware
//
#include <hip/hip_runtime.h>
#include <hip/hip_bf16.h>

typedef __attribute__((ext_vector_type(16))) _Float16 v16h;
typedef __attribute__((ext_vector_type(8)))  _Float16 v8h;
typedef __attribute__((ext_vector_type(8)))  float    v8f;

#define B_  2
#define H_  8
#define S_  2048
#define D_  64
#define QK  6
#define KK  11
#define NEGV (-1.0e9f)

// ---------------------------------------------------------------------------
// Kernel 1: masked depthwise conv (6x11, SAME) + bias + causal re-mask +
//           row softmax, emitting f16 probabilities.
// One block per (b,h,s) query row. 256 threads, 8 columns per thread.
// LDS: 6 input rows with 5-column zero halo on each side (scores are L2
// resident: 256MB f32, only neighbouring rows need to stay in L2, so the 6x re-read across blocks is cheap).
// ---------------------------------------------------------------------------
__global__ __launch_bounds__(256)
void k1_conv_softmax(const float* __restrict__ scores,
                     const float* __restrict__ kq_w,
                     const float* __restrict__ kq_b,
                     _Float16* __restrict__ attn)
{
    const int s = blockIdx.x;
    const int h = blockIdx.y;
    const int b = blockIdx.z;
    const int tid = threadIdx.x;

    __shared__ float sm[QK][S_ + 10];   // +5 halo each side
    __shared__ float wsm[QK * KK];
    __shared__ float red[8];

    if (tid < QK * KK) wsm[tid] = kq_w[h * (QK * KK) + tid];

    const size_t rowBase = ((size_t)b * H_ + h) * (size_t)S_;

    // Load 6 source rows (s-2 .. s+3), causally zeroed, zero-padded halo.
    for (int idx = tid; idx < QK * (S_ + 10); idx += 256) {
        const int r = idx / (S_ + 10);
        const int c = idx - r * (S_ + 10);
        const int sp = s + r - 2;       // pad_low = (6-1)/2 = 2
        const int k  = c - 5;           // pad_low = 5 for the 11-tap dim
        float val = 0.0f;
        if (sp >= 0 && sp < S_ && k >= 0 && k < S_ && k <= sp)
            val = scores[(rowBase + (size_t)sp) * S_ + k];
        sm[r][c] = val;
    }
    __syncthreads();

    const float bias = kq_b[h];
    float vout[8];
    float mx = -3.4e38f;
    #pragma unroll
    for (int i = 0; i < 8; ++i) {
        const int k = tid + 256 * i;
        float acc = bias;
        #pragma unroll
        for (int dq = 0; dq < QK; ++dq) {
            #pragma unroll
            for (int dk = 0; dk < KK; ++dk) {
                // memory column (k + dk - 5) maps to LDS column (k + dk)
                acc = fmaf(wsm[dq * KK + dk], sm[dq][k + dk], acc);
            }
        }
        const float val = (k <= s) ? acc : NEGV;
        vout[i] = val;
        mx = fmaxf(mx, val);
    }

    // block max
    #pragma unroll
    for (int off = 16; off > 0; off >>= 1)
        mx = fmaxf(mx, __shfl_xor(mx, off, 32));
    if ((tid & 31) == 0) red[tid >> 5] = mx;
    __syncthreads();
    mx = red[0];
    #pragma unroll
    for (int i = 1; i < 8; ++i) mx = fmaxf(mx, red[i]);

    // exp + block sum
    float e[8];
    float se = 0.0f;
    #pragma unroll
    for (int i = 0; i < 8; ++i) { e[i] = __expf(vout[i] - mx); se += e[i]; }
    #pragma unroll
    for (int off = 16; off > 0; off >>= 1)
        se += __shfl_xor(se, off, 32);
    __syncthreads();                      // red reuse hazard
    if ((tid & 31) == 0) red[tid >> 5] = se;
    __syncthreads();
    float tot = 0.0f;
    #pragma unroll
    for (int i = 0; i < 8; ++i) tot += red[i];
    const float inv = 1.0f / tot;

    _Float16* orow = attn + (rowBase + (size_t)s) * S_;
    #pragma unroll
    for (int i = 0; i < 8; ++i)
        orow[tid + 256 * i] = (_Float16)(e[i] * inv);
}

// ---------------------------------------------------------------------------
// Kernel 2: transpose + downconvert v [b,h,k,d] f32 -> vt [b,h,d,k] f16 so
// the WMMA B-fragment loads become two contiguous 16B reads per lane.
// ---------------------------------------------------------------------------
__global__ __launch_bounds__(256)
void k2_vt(const float* __restrict__ v, _Float16* __restrict__ vt)
{
    const int bh = blockIdx.x;
    const float* vin = v + (size_t)bh * S_ * D_;
    _Float16* vo = vt + (size_t)bh * D_ * S_;
    for (int idx = threadIdx.x; idx < D_ * S_; idx += 256) {
        const int d = idx >> 11;          // / 2048
        const int k = idx & (S_ - 1);
        vo[idx] = (_Float16)vin[(size_t)k * D_ + d];
    }
}

// ---------------------------------------------------------------------------
// Kernel 3: head-mix (packed f16 FMA) + attn@v via v_wmma_f32_16x16x32_f16 +
// fused LayerNorm over D.
// Block = 128 threads (4 wave32) covering one head-GROUP (both output heads
// j=0 and j=1): the raw attn fragments a0/a1 are shared by both output heads,
// so each pair of A loads feeds TWO WMMAs (two mixes x two V heads) ->
// 4 b128 loads per WMMA and half the attn L2 traffic vs. one-head blocks.
// Each wave owns one 16-column slice of D for both heads; K loop over 2048
// in steps of 32.  A-fragment K packing per ISA: lanes 0-15 -> K {0..7,16..23},
// lanes 16-31 -> K {8..15,24..31}.
// ---------------------------------------------------------------------------
__global__ __launch_bounds__(128)
void k3_mix_gemm_ln(const _Float16* __restrict__ attn,
                    const _Float16* __restrict__ vt,
                    const float* __restrict__ head_mix,
                    const float* __restrict__ gamma,
                    const float* __restrict__ beta,
                    float* __restrict__ out)
{
    const int m0 = blockIdx.x * 16;
    const int g  = blockIdx.y;            // head group (HEAD_K = 2)
    const int b  = blockIdx.z;

    const int lane = threadIdx.x & 31;
    const int w    = threadIdx.x >> 5;

    // mixed_j = M[g,0,j]*attn_{2g} + M[g,1,j]*attn_{2g+1}
    const _Float16 c00 = (_Float16)head_mix[g * 4 + 0];   // i=0, j=0
    const _Float16 c01 = (_Float16)head_mix[g * 4 + 1];   // i=0, j=1
    const _Float16 c10 = (_Float16)head_mix[g * 4 + 2];   // i=1, j=0
    const _Float16 c11 = (_Float16)head_mix[g * 4 + 3];   // i=1, j=1

    const int mrow = m0 + (lane & 15);
    const int koff = (lane >> 4) << 3;    // 0 or 8

    const size_t bh0 = (size_t)b * H_ + (g * 2 + 0);
    const size_t bh1 = (size_t)b * H_ + (g * 2 + 1);
    const _Float16* a0p = attn + (bh0 * S_ + (size_t)mrow) * S_ + koff;
    const _Float16* a1p = attn + (bh1 * S_ + (size_t)mrow) * S_ + koff;

    const int n = 16 * w + (lane & 15);
    const _Float16* b0p = vt + (bh0 * D_ + (size_t)n) * S_ + koff;
    const _Float16* b1p = vt + (bh1 * D_ + (size_t)n) * S_ + koff;

    v8f acc0 = {};
    v8f acc1 = {};
    for (int k0 = 0; k0 < S_; k0 += 32) {
        const v8h a0lo = *(const v8h*)(a0p + k0);
        const v8h a0hi = *(const v8h*)(a0p + k0 + 16);
        const v8h a1lo = *(const v8h*)(a1p + k0);
        const v8h a1hi = *(const v8h*)(a1p + k0 + 16);
        const v8h b0lo = *(const v8h*)(b0p + k0);
        const v8h b0hi = *(const v8h*)(b0p + k0 + 16);
        const v8h b1lo = *(const v8h*)(b1p + k0);
        const v8h b1hi = *(const v8h*)(b1p + k0 + 16);

        // two head-mixes from one pair of A fragments (v_pk_fma_f16)
        const v8h m0lo = a0lo * c00 + a1lo * c10;
        const v8h m0hi = a0hi * c00 + a1hi * c10;
        const v8h m1lo = a0lo * c01 + a1lo * c11;
        const v8h m1hi = a0hi * c01 + a1hi * c11;

        const v16h A0 = __builtin_shufflevector(m0lo, m0hi,
            0,1,2,3,4,5,6,7,8,9,10,11,12,13,14,15);
        const v16h A1 = __builtin_shufflevector(m1lo, m1hi,
            0,1,2,3,4,5,6,7,8,9,10,11,12,13,14,15);
        const v16h B0 = __builtin_shufflevector(b0lo, b0hi,
            0,1,2,3,4,5,6,7,8,9,10,11,12,13,14,15);
        const v16h B1 = __builtin_shufflevector(b1lo, b1hi,
            0,1,2,3,4,5,6,7,8,9,10,11,12,13,14,15);

        acc0 = __builtin_amdgcn_wmma_f32_16x16x32_f16(
            false, A0, false, B0, (short)0, acc0, false, false);
        acc1 = __builtin_amdgcn_wmma_f32_16x16x32_f16(
            false, A1, false, B1, (short)0, acc1, false, false);
    }

    // Scatter C tiles to LDS: VGPR r -> row r (lanes 0-15) / r+8 (lanes 16-31)
    __shared__ float tile[2][16][72];
    const int rbase = (lane >> 4) << 3;
    const int col   = 16 * w + (lane & 15);
    #pragma unroll
    for (int r = 0; r < 8; ++r) {
        tile[0][rbase + r][col] = acc0[r];
        tile[1][rbase + r][col] = acc1[r];
    }
    __syncthreads();

    // LayerNorm over D=64 for 32 rows (2 heads x 16): 4 threads per row,
    // 16 cols each, shuffle-reduce within aligned 4-lane groups.
    const int t  = threadIdx.x;
    const int jj = t >> 6;                // output head within group
    const int r  = (t >> 2) & 15;        // row within tile
    const int cb = (t & 3) * 16;         // column base
    float x[16];
    float sum = 0.0f, sq = 0.0f;
    #pragma unroll
    for (int i = 0; i < 16; ++i) {
        x[i] = tile[jj][r][cb + i];
        sum += x[i];
        sq  += x[i] * x[i];
    }
    #pragma unroll
    for (int off = 1; off < 4; off <<= 1) {
        sum += __shfl_xor(sum, off, 32);
        sq  += __shfl_xor(sq,  off, 32);
    }
    const float mean = sum * (1.0f / 64.0f);
    const float var  = sq * (1.0f / 64.0f) - mean * mean;
    const float rs   = rsqrtf(var + 1e-5f);

    const size_t obase =
        (((size_t)b * H_ + (g * 2 + jj)) * S_ + (size_t)(m0 + r)) * D_ + cb;
    #pragma unroll
    for (int i = 0; i < 16; ++i)
        out[obase + i] = (x[i] - mean) * rs * gamma[cb + i] + beta[cb + i];
}

// ---------------------------------------------------------------------------
extern "C" void kernel_launch(void* const* d_in, const int* in_sizes, int n_in,
                              void* d_out, int out_size, void* d_ws, size_t ws_size,
                              hipStream_t stream)
{
    const float* scores   = (const float*)d_in[0];
    const float* v        = (const float*)d_in[1];
    const float* kq_w     = (const float*)d_in[2];
    const float* kq_b     = (const float*)d_in[3];
    const float* head_mix = (const float*)d_in[4];
    const float* gamma    = (const float*)d_in[5];
    const float* beta     = (const float*)d_in[6];
    float* out = (float*)d_out;

    // workspace: [attn f16 : B*H*S*S] [vt f16 : B*H*D*S]  (128MB + 4MB)
    _Float16* wsAttn = (_Float16*)d_ws;
    _Float16* wsVt   = (_Float16*)((char*)d_ws +
                        (size_t)B_ * H_ * S_ * S_ * sizeof(_Float16));

    dim3 g1(S_, H_, B_);
    k1_conv_softmax<<<g1, 256, 0, stream>>>(scores, kq_w, kq_b, wsAttn);

    k2_vt<<<B_ * H_, 256, 0, stream>>>(v, wsVt);

    dim3 g3(S_ / 16, H_ / 2, B_);   // one block per head GROUP
    k3_mix_gemm_ln<<<g3, 128, 0, stream>>>(wsAttn, wsVt, head_mix,
                                           gamma, beta, out);
}
